// GNNEmbedder_Conv_27419071217742
// MI455X (gfx1250) — matrix operand order repaired and checked by reading, not compiled, not run on hardware
//
#include <hip/hip_runtime.h>
#include <hip/hip_bf16.h>

typedef __attribute__((ext_vector_type(16))) _Float16 v16h;
typedef __attribute__((ext_vector_type(8)))  float    v8f;
typedef __attribute__((ext_vector_type(4)))  float    v4f;

#define N_NODES 50000
#define N_EDGES 250000
#define N_GRAPHS 512
#define F_IN 32
#define F_EDGE 16
#define LATENT 32
#define EMBED 128

// workspace layout (bytes)
#define OFF_H    0u                       // E*32 f16      = 16,000,000
#define OFF_AGG  16000000u                // N*32 f32      =  6,400,000
#define OFF_SUMS 22400000u                // 512*32 f32    =     65,536
#define OFF_CNTS 22465536u                // 512 f32       =      2,048
#define OFF_T    22467584u                // N*32 f32      =  6,400,000
#define ZERO_FLOATS ((N_NODES*32 + N_GRAPHS*32 + N_GRAPHS))   // agg..cnts contiguous

// ---------------- kernel 0: zero agg/sums/cnts ----------------
__global__ void k0_zero(float* __restrict__ p, int n) {
    int t = blockIdx.x * blockDim.x + threadIdx.x;
    if (t < n) p[t] = 0.0f;
}

// ---------------- kernel 1: edge MLP  h = relu(ea@W1+b1) -> f16 ----------------
__global__ void k1_edge_mlp(const float* __restrict__ ea, const float* __restrict__ W1,
                            const float* __restrict__ b1, _Float16* __restrict__ h) {
    int t = blockIdx.x * blockDim.x + threadIdx.x;
    if (t >= N_EDGES * LATENT) return;
    int e = t >> 5, k = t & 31;
    float acc = b1[k];
    const float* er = ea + e * F_EDGE;
    #pragma unroll
    for (int j = 0; j < F_EDGE; ++j) acc += er[j] * W1[j * LATENT + k];
    h[t] = (_Float16)(acc > 0.f ? acc : 0.f);
}

// ---------------- kernel 1b: t[n][l] = sum_f x[n][f]*b2[f*32+l] ----------------
__global__ void k1b_bias(const float* __restrict__ x, const float* __restrict__ b2,
                         float* __restrict__ tb) {
    int t = blockIdx.x * blockDim.x + threadIdx.x;
    if (t >= N_NODES * LATENT) return;
    int n = t >> 5, l = t & 31;
    float acc = 0.f;
    const float* xr = x + n * F_IN;
    #pragma unroll
    for (int f = 0; f < F_IN; ++f) acc += xr[f] * b2[f * LATENT + l];
    tb[t] = acc;
}

// Per-tile A operands: x fragment (f16 A layout) + full h row (2x v16h)
struct TileA { v16h xf; v16h h0; v16h h1; };

__device__ __forceinline__ TileA load_tileA(const float* __restrict__ x,
                                            const int* __restrict__ edge_index,
                                            const _Float16* __restrict__ h,
                                            int eBase, int lane, int fbase) {
    TileA r;
    int e = eBase + (lane & 15);
    int src = edge_index[e];                           // src row of edge_index
    const float* xr = x + (size_t)src * F_IN;
    // ISA 16-bit A layout: lane<16: f=0..7,16..23 ; lane>=16: f=8..15,24..31
    v4f xa0 = *(const v4f*)(xr + fbase);
    v4f xa1 = *(const v4f*)(xr + fbase + 4);
    v4f xb0 = *(const v4f*)(xr + fbase + 16);
    v4f xb1 = *(const v4f*)(xr + fbase + 20);
    #pragma unroll
    for (int j = 0; j < 4; ++j) {
        r.xf[j]      = (_Float16)xa0[j];
        r.xf[4 + j]  = (_Float16)xa1[j];
        r.xf[8 + j]  = (_Float16)xb0[j];
        r.xf[12 + j] = (_Float16)xb1[j];
    }
    const v16h* hp = (const v16h*)(h + (size_t)e * LATENT);   // 64B row, 64B aligned
    r.h0 = hp[0];
    r.h1 = hp[1];
    return r;
}

__device__ __forceinline__ void scatter_tile(const int* __restrict__ edge_index,
                                             const float* __restrict__ tb,
                                             float* __restrict__ agg,
                                             int eBase, int lane,
                                             const v8f& acc0, const v8f& acc1) {
    // C layout: VGPR r holds M = r + 8*(lane>=16), column l = (lane&15) (+16 for acc1)
    int l0   = lane & 15;
    int moff = (lane >> 4) << 3;
    #pragma unroll
    for (int r = 0; r < 8; ++r) {
        int er  = eBase + moff + r;
        int s_r = edge_index[er];                  // src for b2 bias term
        int d   = edge_index[N_EDGES + er];        // dst (row 1)
        float v0 = acc0[r] + tb[(size_t)s_r * LATENT + l0];
        float v1 = acc1[r] + tb[(size_t)s_r * LATENT + 16 + l0];
        atomicAdd(&agg[(size_t)d * LATENT + l0], v0);
        atomicAdd(&agg[(size_t)d * LATENT + 16 + l0], v1);
    }
}

// ---------------- kernel 2: fused msg GEMM + scatter-add (the big one) ----------------
// msg = Z @ W2' with Z[e, k*32+f] = h[e][k] * x[src_e][f]  (A built on the fly)
// Two 16-edge tiles per iteration so each LDS B fragment feeds 2 WMMAs.
__global__ __launch_bounds__(256) void k2_edge_gemm(
    const float* __restrict__ x, const int* __restrict__ edge_index,
    const _Float16* __restrict__ h, const float* __restrict__ W2,
    const float* __restrict__ tb, float* __restrict__ agg,
    int numTiles, int totalWaves)
{
    // B fragments: [chunk kk 0..31][nhalf 0..1][lane 0..31][16 halves] = 64 KB
    __shared__ _Float16 Bp[32 * 2 * 32 * 16];
    for (int idx = threadIdx.x; idx < 32 * 2 * 32 * 16; idx += blockDim.x) {
        int i    = idx & 15;
        int lane = (idx >> 4) & 31;
        int nh   = (idx >> 9) & 1;
        int kk   = idx >> 10;
        int l = (lane & 15) + nh * 16;        // column N
        int K = ((i & 8) << 1) + ((lane >> 4) << 3) + (i & 7);       // row K within chunk (f index)
        Bp[idx] = (_Float16)W2[kk * 1024 + K * LATENT + l];
    }
    __syncthreads();

    const int lane  = threadIdx.x & 31;
    const int wave  = (blockIdx.x * blockDim.x + threadIdx.x) >> 5;
    const int fbase = (lane >> 4) << 3;       // 0 or 8
    const int numPairs = (numTiles + 1) >> 1;

    for (int p = wave; p < numPairs; p += totalWaves) {
        int tile0 = p * 2;
        int tile1 = tile0 + 1;
        bool has1 = (tile1 < numTiles);
        int tile1c = has1 ? tile1 : tile0;    // clamp: compute is safe, scatter masked

        // Fence: keeps the Bp ds_loads inside the loop (prevents LICM -> scratch spills)
        asm volatile("" ::: "memory");

        int eBase0 = tile0 << 4;
        int eBase1 = tile1c << 4;
        TileA t0 = load_tileA(x, edge_index, h, eBase0, lane, fbase);
        TileA t1 = load_tileA(x, edge_index, h, eBase1, lane, fbase);

        // prefetch next pair's h rows
        int np = p + totalWaves;
        if (np < numPairs)
            __builtin_prefetch(h + (size_t)((np << 5) + (lane & 15)) * LATENT, 0, 1);

        v8f acc00 = {}, acc01 = {}, acc10 = {}, acc11 = {};
        #pragma unroll
        for (int kk = 0; kk < 32; ++kk) {
            v16h b0 = *(const v16h*)(Bp + (((kk * 2 + 0) * 32) + lane) * 16);
            v16h b1 = *(const v16h*)(Bp + (((kk * 2 + 1) * 32) + lane) * 16);
            _Float16 hk0 = (kk < 16) ? t0.h0[kk] : t0.h1[kk - 16];
            _Float16 hk1 = (kk < 16) ? t1.h0[kk] : t1.h1[kk - 16];
            v16h a0, a1;
            #pragma unroll
            for (int j = 0; j < 16; ++j) { a0[j] = t0.xf[j] * hk0; a1[j] = t1.xf[j] * hk1; }
            acc00 = __builtin_amdgcn_wmma_f32_16x16x32_f16(false, a0, false, b0, (short)0, acc00, false, false);
            acc01 = __builtin_amdgcn_wmma_f32_16x16x32_f16(false, a0, false, b1, (short)0, acc01, false, false);
            acc10 = __builtin_amdgcn_wmma_f32_16x16x32_f16(false, a1, false, b0, (short)0, acc10, false, false);
            acc11 = __builtin_amdgcn_wmma_f32_16x16x32_f16(false, a1, false, b1, (short)0, acc11, false, false);
        }

        scatter_tile(edge_index, tb, agg, eBase0, lane, acc00, acc01);
        if (has1)
            scatter_tile(edge_index, tb, agg, eBase1, lane, acc10, acc11);
    }
}

// ---------------- kernel 3: node = relu(agg + x@root + bias), fused mean-pool scatter ----------------
__global__ __launch_bounds__(256) void k3_node_pool(
    const float* __restrict__ x, const float* __restrict__ root,
    const float* __restrict__ conv_bias, const int* __restrict__ batch,
    const float* __restrict__ agg, float* __restrict__ sums,
    float* __restrict__ cnts, int numTiles, int totalWaves)
{
    __shared__ _Float16 Rp[2 * 32 * 16];   // root as B fragments, 2 KB
    for (int idx = threadIdx.x; idx < 2 * 32 * 16; idx += blockDim.x) {
        int i = idx & 15, lane = (idx >> 4) & 31, nh = (idx >> 9) & 1;
        int l = (lane & 15) + nh * 16;
        int K = ((i & 8) << 1) + ((lane >> 4) << 3) + (i & 7);
        Rp[idx] = (_Float16)root[K * LATENT + l];
    }
    __syncthreads();

    const int lane  = threadIdx.x & 31;
    const int wave  = (blockIdx.x * blockDim.x + threadIdx.x) >> 5;
    const int fbase = (lane >> 4) << 3;
    v16h rb0 = *(const v16h*)(Rp + (0 * 32 + lane) * 16);
    v16h rb1 = *(const v16h*)(Rp + (1 * 32 + lane) * 16);

    for (int tile = wave; tile < numTiles; tile += totalWaves) {
        int nBase = tile << 4;
        const float* xr = x + (size_t)(nBase + (lane & 15)) * F_IN;
        v4f xa0 = *(const v4f*)(xr + fbase);
        v4f xa1 = *(const v4f*)(xr + fbase + 4);
        v4f xb0 = *(const v4f*)(xr + fbase + 16);
        v4f xb1 = *(const v4f*)(xr + fbase + 20);
        v16h xf;
        #pragma unroll
        for (int j = 0; j < 4; ++j) {
            xf[j]      = (_Float16)xa0[j];
            xf[4 + j]  = (_Float16)xa1[j];
            xf[8 + j]  = (_Float16)xb0[j];
            xf[12 + j] = (_Float16)xb1[j];
        }
        v8f acc0 = {}, acc1 = {};
        acc0 = __builtin_amdgcn_wmma_f32_16x16x32_f16(false, xf, false, rb0, (short)0, acc0, false, false);
        acc1 = __builtin_amdgcn_wmma_f32_16x16x32_f16(false, xf, false, rb1, (short)0, acc1, false, false);

        int l0   = lane & 15;
        int moff = (lane >> 4) << 3;
        #pragma unroll
        for (int r = 0; r < 8; ++r) {
            int n = nBase + moff + r;
            int g = batch[n];
            float v0 = acc0[r] + agg[(size_t)n * LATENT + l0]      + conv_bias[l0];
            float v1 = acc1[r] + agg[(size_t)n * LATENT + 16 + l0] + conv_bias[16 + l0];
            v0 = v0 > 0.f ? v0 : 0.f;
            v1 = v1 > 0.f ? v1 : 0.f;
            atomicAdd(&sums[(size_t)g * LATENT + l0], v0);
            atomicAdd(&sums[(size_t)g * LATENT + 16 + l0], v1);
            if (l0 == 0) atomicAdd(&cnts[g], 1.0f);   // lanes 0 & 16 cover M=0..7 / 8..15
        }
    }
}

// ---------------- kernel 4: out = relu(pooled) @ fc_w + fc_b ----------------
__global__ void k4_fc(const float* __restrict__ sums, const float* __restrict__ cnts,
                      const float* __restrict__ fc_w, const float* __restrict__ fc_b,
                      float* __restrict__ out) {
    int t = blockIdx.x * blockDim.x + threadIdx.x;
    if (t >= N_GRAPHS * EMBED) return;
    int g = t >> 7, o = t & 127;
    float c = cnts[g]; c = c > 1.f ? c : 1.f;
    float acc = fc_b[o];
    #pragma unroll
    for (int l = 0; l < LATENT; ++l) {
        float p = sums[g * LATENT + l] / c;
        p = p > 0.f ? p : 0.f;
        acc += p * fc_w[l * EMBED + o];
    }
    out[t] = acc;
}

extern "C" void kernel_launch(void* const* d_in, const int* in_sizes, int n_in,
                              void* d_out, int out_size, void* d_ws, size_t ws_size,
                              hipStream_t stream) {
    const float* x         = (const float*)d_in[0];
    const int*   edge_idx  = (const int*)d_in[1];
    const float* edge_attr = (const float*)d_in[2];
    const int*   batch     = (const int*)d_in[3];
    const float* W1        = (const float*)d_in[4];
    const float* b1        = (const float*)d_in[5];
    const float* W2        = (const float*)d_in[6];
    const float* b2        = (const float*)d_in[7];
    const float* root      = (const float*)d_in[8];
    const float* conv_bias = (const float*)d_in[9];
    const float* fc_w      = (const float*)d_in[10];
    const float* fc_b      = (const float*)d_in[11];
    float* out = (float*)d_out;

    char* ws = (char*)d_ws;
    _Float16* h    = (_Float16*)(ws + OFF_H);
    float*    agg  = (float*)(ws + OFF_AGG);
    float*    sums = (float*)(ws + OFF_SUMS);
    float*    cnts = (float*)(ws + OFF_CNTS);
    float*    tb   = (float*)(ws + OFF_T);

    // 0) zero agg/sums/cnts (contiguous)
    int zn = ZERO_FLOATS;
    k0_zero<<<(zn + 255) / 256, 256, 0, stream>>>(agg, zn);
    // 1) edge MLP -> h (f16)
    k1_edge_mlp<<<(N_EDGES * LATENT + 255) / 256, 256, 0, stream>>>(edge_attr, W1, b1, h);
    // 1b) per-node b2 bias term
    k1b_bias<<<(N_NODES * LATENT + 255) / 256, 256, 0, stream>>>(x, b2, tb);
    // 2) fused edge GEMM + scatter (2 tiles per wave-iteration)
    {
        int blocks = 240, waves = blocks * (256 / 32);
        k2_edge_gemm<<<blocks, 256, 0, stream>>>(x, edge_idx, h, W2, tb, agg,
                                                 N_EDGES / 16, waves);
    }
    // 3) node update + pooling
    {
        int blocks = 120, waves = blocks * (256 / 32);
        k3_node_pool<<<blocks, 256, 0, stream>>>(x, root, conv_bias, batch, agg,
                                                 sums, cnts, N_NODES / 16, waves);
    }
    // 4) final FC
    k4_fc<<<(N_GRAPHS * EMBED + 255) / 256, 256, 0, stream>>>(sums, cnts, fc_w, fc_b, out);
}
